// SampleNetBatchNorm_87866440941655
// MI455X (gfx1250) — compile-verified, matrix operand order checked by reading, not run on hardware
//
#include <hip/hip_runtime.h>

// ---------------------------------------------------------------------------
// MI455X (gfx1250) implementation: bf16 WMMA MLP pipeline with on-device
// BatchNorm folding.  Wave32, v_wmma_f32_16x16x32_bf16 as the matmul engine.
// ---------------------------------------------------------------------------

typedef __attribute__((ext_vector_type(16))) __bf16    v16bf;
typedef __attribute__((ext_vector_type(2)))  __bf16    v2bf;
typedef __attribute__((ext_vector_type(8)))  float     v8f;
typedef __attribute__((ext_vector_type(4)))  float     f32x4;
typedef __attribute__((ext_vector_type(4)))  unsigned  u32x4;

#define DEV __device__ __forceinline__

// fp32 -> bf16 via the compiler's native fptrunc (round-nearest-even); the
// backend selects the hardware convert instruction when available.
DEV unsigned short f2bf(float f) {
  __bf16 b = (__bf16)f;
  return __builtin_bit_cast(unsigned short, b);
}
DEV unsigned pk2bf(float a, float b) {
  v2bf r = { (__bf16)a, (__bf16)b };
  return __builtin_bit_cast(unsigned, r);
}

DEV float bf2f(unsigned short h) { return __uint_as_float(((unsigned)h) << 16); }
DEV float eluf(float x) { return x > 0.f ? x : __expf(x) - 1.f; }

union VB {
  v16bf v;
  unsigned short h[16];
  unsigned u[8];
  u32x4 q[2];
};

// ---------------------------------------------------------------------------
// Generic fused layer:  Y = elu( (X*scale+shift) @ W^T + bias )
//  - scale/shift (BN of the *previous* layer) are folded into W / bias.
//  - writes Y as bf16 (32 cols), or (SEG) atomically segment-sums fp32 Y.
//  - accumulates per-feature sum / sum^2 of Y into stat[0..63].
// One wave computes a 16x32 tile (two v_wmma_f32_16x16x32_bf16 per K-step).
// ---------------------------------------------------------------------------
template<int KDIM, bool IN_F32, bool SEG>
__global__ __launch_bounds__(256)
void layer_kernel(const void* __restrict__ inp, int ldin, int rows, int tiles,
                  const float* __restrict__ W, int ldw,
                  const float* __restrict__ bias,
                  const float* __restrict__ scl, const float* __restrict__ shf,
                  void* __restrict__ outp,
                  const int* __restrict__ segid,
                  float* __restrict__ stat)
{
  constexpr int KS = KDIM / 32;
  __shared__ float ssum[32], ssq[32];
  const int tid = threadIdx.x;
  if (tid < 32) { ssum[tid] = 0.f; ssq[tid] = 0.f; }
  __syncthreads();

  const int lane = tid & 31;
  const int wave = tid >> 5;
  const int hi   = lane >> 4;   // which half of the wave (K-interleave)
  const int hl   = lane & 15;   // row (A) or column (B/C)

  // ---- build folded B operands (weights) + folded bias, once per wave ----
  VB Bv[2][KS];
  float bfold[2];
#pragma unroll
  for (int t = 0; t < 2; ++t) {
    const int o = hl + 16 * t;                   // output column
    float bb = bias[o];
    if (shf) {
      for (int i = 0; i < ldw; ++i) bb += W[o * ldw + i] * shf[i];
    }
    bfold[t] = bb;
#pragma unroll
    for (int s = 0; s < KS; ++s) {
#pragma unroll
      for (int e = 0; e < 16; e += 2) {
        const int k = s * 32 + (e & 7) + ((e >= 8) ? 16 : 0) + (hi ? 8 : 0);
        float w0 = 0.f, w1 = 0.f;
        if (k < ldw) {
          w0 = W[o * ldw + k];
          w1 = W[o * ldw + k + 1];
          if (scl) { w0 *= scl[k]; w1 *= scl[k + 1]; }
        }
        Bv[t][s].u[e >> 1] = pk2bf(w0, w1);
      }
    }
  }

  float ps0 = 0.f, pq0 = 0.f, ps1 = 0.f, pq1 = 0.f;

  for (int it = 0; it < tiles; ++it) {
    const int rowBase = (blockIdx.x * tiles + it) * 128 + wave * 16;
    const int row = rowBase + hl;

    v8f acc0 = {0.f,0.f,0.f,0.f,0.f,0.f,0.f,0.f};
    v8f acc1 = {0.f,0.f,0.f,0.f,0.f,0.f,0.f,0.f};

#pragma unroll
    for (int s = 0; s < KS; ++s) {
      VB A;
      if (IN_F32) {
        const float* rp = (const float*)inp + (size_t)row * ldin;
        const int k0 = s * 32 + hi * 8;
        const int k1 = s * 32 + 16 + hi * 8;
        const f32x4 z = {0.f, 0.f, 0.f, 0.f};
        f32x4 c0 = (k0 < ldin) ? __builtin_nontemporal_load((const f32x4*)(rp + k0))     : z;
        f32x4 c1 = (k0 < ldin) ? __builtin_nontemporal_load((const f32x4*)(rp + k0 + 4)) : z;
        f32x4 c2 = (k1 < ldin) ? __builtin_nontemporal_load((const f32x4*)(rp + k1))     : z;
        f32x4 c3 = (k1 < ldin) ? __builtin_nontemporal_load((const f32x4*)(rp + k1 + 4)) : z;
        A.u[0] = pk2bf(c0.x, c0.y); A.u[1] = pk2bf(c0.z, c0.w);
        A.u[2] = pk2bf(c1.x, c1.y); A.u[3] = pk2bf(c1.z, c1.w);
        A.u[4] = pk2bf(c2.x, c2.y); A.u[5] = pk2bf(c2.z, c2.w);
        A.u[6] = pk2bf(c3.x, c3.y); A.u[7] = pk2bf(c3.z, c3.w);
      } else {
        const u32x4* rp = (const u32x4*)((const unsigned short*)inp + (size_t)row * ldin);
        A.q[0] = __builtin_nontemporal_load(rp + s * 4 + hi);
        A.q[1] = __builtin_nontemporal_load(rp + s * 4 + hi + 2);
      }
      acc0 = __builtin_amdgcn_wmma_f32_16x16x32_bf16(false, A.v, false, Bv[0][s].v,
                                                     (short)0, acc0, false, false);
      acc1 = __builtin_amdgcn_wmma_f32_16x16x32_bf16(false, A.v, false, Bv[1][s].v,
                                                     (short)0, acc1, false, false);
    }

    float y0[8], y1[8];
#pragma unroll
    for (int r = 0; r < 8; ++r) {
      float a = eluf(acc0[r] + bfold[0]);
      float b = eluf(acc1[r] + bfold[1]);
      y0[r] = a; y1[r] = b;
      ps0 += a; pq0 += a * a;
      ps1 += b; pq1 += b * b;
    }

    if (SEG) {
      float* seg = (float*)outp;
      const int* sg = segid + rowBase + hi * 8;
#pragma unroll
      for (int r = 0; r < 8; ++r) {
        float* p = seg + (size_t)sg[r] * 32 + hl;
        atomicAdd(p,      y0[r]);
        atomicAdd(p + 16, y1[r]);
      }
    } else {
      // single base pointer; per-store offsets are compile-time constants so
      // they fold into the global_store_b16 immediate offset field
      unsigned short* p = (unsigned short*)outp + (size_t)(rowBase + hi * 8) * 32 + hl;
#pragma unroll
      for (int r = 0; r < 8; ++r) {
        p[r * 32]      = f2bf(y0[r]);
        p[r * 32 + 16] = f2bf(y1[r]);
      }
    }
  }

  atomicAdd(&ssum[hl],      ps0);
  atomicAdd(&ssq [hl],      pq0);
  atomicAdd(&ssum[hl + 16], ps1);
  atomicAdd(&ssq [hl + 16], pq1);
  __syncthreads();
  if (tid < 32) {
    atomicAdd(&stat[tid],      ssum[tid]);
    atomicAdd(&stat[tid + 32], ssq [tid]);
  }
}

// stats -> (scale, shift) for the next layer's weight fold:
//   scale = g/sqrt(var+eps), shift = be - mean*scale
__global__ void finalize_kernel(const float* __restrict__ stat,
                                const float* __restrict__ g,
                                const float* __restrict__ be,
                                float n,
                                float* __restrict__ scl, float* __restrict__ shf)
{
  int i = threadIdx.x;
  if (i >= 32) return;
  float m   = stat[i] / n;
  float var = stat[i + 32] / n - m * m;
  float sc  = rsqrtf(var + 1e-5f) * g[i];
  scl[i] = sc;
  shf[i] = be[i] - m * sc;
}

__global__ void count_kernel(const int* __restrict__ ids, float* __restrict__ cnt, int n)
{
  int t = blockIdx.x * blockDim.x + threadIdx.x;
  if (t < n) atomicAdd(&cnt[ids[t]], 1.f);
}

__global__ void featsum_kernel(const float* __restrict__ feat, const int* __restrict__ ids,
                               float* __restrict__ fsum, int M, int C)
{
  int t = blockIdx.x * blockDim.x + threadIdx.x;
  if (t < M * C) {
    int m = t / C, c = t - m * C;
    atomicAdd(&fsum[(size_t)ids[m] * C + c], feat[t]);
  }
}

__global__ void featdiv_kernel(float* __restrict__ f, const float* __restrict__ cnt,
                               int Bn, int C)
{
  int t = blockIdx.x * blockDim.x + threadIdx.x;
  if (t < Bn * C) f[t] /= fmaxf(cnt[t / C], 1.f);
}

// seg_sum(BN(y)) = sc*Sum(y) + cnt_s*sh  -> materialize BN'd segment features
__global__ void segbn_kernel(const float* __restrict__ seg, const float* __restrict__ cnt,
                             const float* __restrict__ scl, const float* __restrict__ shf,
                             unsigned short* __restrict__ xs, int Bn)
{
  int t = blockIdx.x * blockDim.x + threadIdx.x;
  if (t < Bn * 32) {
    int s = t >> 5, i = t & 31;
    xs[t] = f2bf(seg[t] * scl[i] + cnt[s] * shf[i]);
  }
}

// concat( BN(feat_raw), BN(graph_raw) ) -> bf16 [B,64]
__global__ void concat_kernel(const unsigned short* __restrict__ Fm,
                              const unsigned short* __restrict__ Gm,
                              const float* __restrict__ sF, const float* __restrict__ hF,
                              const float* __restrict__ sG, const float* __restrict__ hG,
                              unsigned short* __restrict__ D, int Bn)
{
  int t = blockIdx.x * blockDim.x + threadIdx.x;
  if (t < Bn * 64) {
    int r = t >> 6, c = t & 63;
    float v;
    if (c < 32) v = bf2f(Fm[r * 32 + c]) * sF[c] + hF[c];
    else { int cc = c - 32; v = bf2f(Gm[r * 32 + cc]) * sG[cc] + hG[cc]; }
    D[t] = f2bf(v);
  }
}

// sigmoid( BN(h) @ Wo^T + bo )
__global__ void final_kernel(const unsigned short* __restrict__ H,
                             const float* __restrict__ scl, const float* __restrict__ shf,
                             const float* __restrict__ Wo, const float* __restrict__ bo,
                             float* __restrict__ out, int Bn, int NL)
{
  int t = blockIdx.x * blockDim.x + threadIdx.x;
  if (t < Bn * NL) {
    int r = t / NL, o = t - r * NL;
    float s = bo[o];
    for (int i = 0; i < 32; ++i)
      s += (bf2f(H[r * 32 + i]) * scl[i] + shf[i]) * Wo[o * 32 + i];
    out[t] = 1.f / (1.f + __expf(-s));
  }
}

// ---------------------------------------------------------------------------
extern "C" void kernel_launch(void* const* d_in, const int* in_sizes, int n_in,
                              void* d_out, int out_size, void* d_ws, size_t ws_size,
                              hipStream_t stream)
{
  (void)n_in; (void)ws_size;
  const float* nodeF  = (const float*)d_in[1];   // [N,64]
  const int*   nodeId = (const int*)  d_in[2];   // [N]
  const float* sampF  = (const float*)d_in[3];   // [M,NSF]
  const int*   sampId = (const int*)  d_in[4];   // [M]

  const int N   = in_sizes[2];
  const int M   = in_sizes[4];
  const int NSF = in_sizes[3] / M;               // 200
  const int NL  = in_sizes[47];                  // 2
  const int Bn  = out_size / NL;                 // 4096

  // parameter lists (W, b, g, be) x2 per mlp
  const float* p_nn[8];   for (int i = 0; i < 8; ++i) p_nn[i]   = (const float*)d_in[6  + i];
  const float* p_g1[8];   for (int i = 0; i < 8; ++i) p_g1[i]   = (const float*)d_in[14 + i];
  const float* p_g2[8];   for (int i = 0; i < 8; ++i) p_g2[i]   = (const float*)d_in[22 + i];
  const float* p_ft[8];   for (int i = 0; i < 8; ++i) p_ft[i]   = (const float*)d_in[30 + i];
  const float* p_ot[8];   for (int i = 0; i < 8; ++i) p_ot[i]   = (const float*)d_in[38 + i];
  const float* Wo = (const float*)d_in[46];
  const float* bo = (const float*)d_in[47];

  // -------- workspace carve-up (256B aligned) --------
  char* base = (char*)d_ws;
  size_t cur = 0;
  auto alloc = [&](size_t bytes) -> char* {
    char* p = base + cur;
    cur = (cur + bytes + 255) & ~(size_t)255;
    return p;
  };
  float* stats = (float*)alloc((size_t)10 * 64 * 4);      // zeroed
  float* cntF  = (float*)alloc((size_t)Bn * 4);           // zeroed
  float* cntN  = (float*)alloc((size_t)Bn * 4);           // zeroed
  float* segR  = (float*)alloc((size_t)Bn * 32 * 4);      // zeroed
  float* fmean = (float*)alloc((size_t)Bn * NSF * 4);     // zeroed
  const size_t zeroBytes = cur;
  float* ss    = (float*)alloc((size_t)10 * 64 * 4);      // scale/shift per layer
  unsigned short* B1 = (unsigned short*)alloc((size_t)Bn * 32 * 2);
  unsigned short* B2 = (unsigned short*)alloc((size_t)Bn * 32 * 2);
  unsigned short* C1 = (unsigned short*)alloc((size_t)Bn * 32 * 2);
  unsigned short* C2 = (unsigned short*)alloc((size_t)Bn * 32 * 2);
  unsigned short* XS = (unsigned short*)alloc((size_t)Bn * 32 * 2);
  unsigned short* D0 = (unsigned short*)alloc((size_t)Bn * 64 * 2);
  unsigned short* D1 = (unsigned short*)alloc((size_t)Bn * 32 * 2);
  unsigned short* D2 = (unsigned short*)alloc((size_t)Bn * 32 * 2);
  unsigned short* bufA = (unsigned short*)alloc((size_t)N * 32 * 2);
  unsigned short* bufB = (unsigned short*)alloc((size_t)N * 32 * 2);

  auto ST = [&](int L) { return stats + L * 64; };
  auto SC = [&](int L) { return ss + L * 64; };
  auto SH = [&](int L) { return ss + L * 64 + 32; };

  (void)hipMemsetAsync(d_ws, 0, zeroBytes, stream);

  const int TB = 8;                              // row-tiles per block (big layers)
  const int nbBig   = N / (128 * TB);            // 2000 blocks
  const int nbSmall = Bn / 128;                  // 32 blocks
  const float fN = (float)N, fB = (float)Bn;

  // ---- node path over N rows ----
  layer_kernel<64, true,  false><<<nbBig, 256, 0, stream>>>(
      nodeF, 64, N, TB, p_nn[0], 64, p_nn[1], nullptr, nullptr, bufA, nullptr, ST(0));
  finalize_kernel<<<1, 32, 0, stream>>>(ST(0), p_nn[2], p_nn[3], fN, SC(0), SH(0));

  layer_kernel<32, false, false><<<nbBig, 256, 0, stream>>>(
      bufA, 32, N, TB, p_nn[4], 32, p_nn[5], SC(0), SH(0), bufB, nullptr, ST(1));
  finalize_kernel<<<1, 32, 0, stream>>>(ST(1), p_nn[6], p_nn[7], fN, SC(1), SH(1));

  layer_kernel<32, false, false><<<nbBig, 256, 0, stream>>>(
      bufB, 32, N, TB, p_g1[0], 32, p_g1[1], SC(1), SH(1), bufA, nullptr, ST(2));
  finalize_kernel<<<1, 32, 0, stream>>>(ST(2), p_g1[2], p_g1[3], fN, SC(2), SH(2));

  count_kernel<<<(N + 255) / 256, 256, 0, stream>>>(nodeId, cntN, N);

  layer_kernel<32, false, true><<<nbBig, 256, 0, stream>>>(
      bufA, 32, N, TB, p_g1[4], 32, p_g1[5], SC(2), SH(2), segR, nodeId, ST(3));
  finalize_kernel<<<1, 32, 0, stream>>>(ST(3), p_g1[6], p_g1[7], fN, SC(3), SH(3));

  segbn_kernel<<<(Bn * 32 + 255) / 256, 256, 0, stream>>>(segR, cntN, SC(3), SH(3), XS, Bn);

  // ---- feature path: segment mean then MLP ----
  count_kernel<<<(M + 255) / 256, 256, 0, stream>>>(sampId, cntF, M);
  featsum_kernel<<<(M * NSF + 255) / 256, 256, 0, stream>>>(sampF, sampId, fmean, M, NSF);
  featdiv_kernel<<<(Bn * NSF + 255) / 256, 256, 0, stream>>>(fmean, cntF, Bn, NSF);

  layer_kernel<224, true, false><<<nbSmall, 256, 0, stream>>>(
      fmean, NSF, Bn, 1, p_ft[0], NSF, p_ft[1], nullptr, nullptr, B1, nullptr, ST(4));
  finalize_kernel<<<1, 32, 0, stream>>>(ST(4), p_ft[2], p_ft[3], fB, SC(4), SH(4));

  layer_kernel<32, false, false><<<nbSmall, 256, 0, stream>>>(
      B1, 32, Bn, 1, p_ft[4], 32, p_ft[5], SC(4), SH(4), B2, nullptr, ST(5));
  finalize_kernel<<<1, 32, 0, stream>>>(ST(5), p_ft[6], p_ft[7], fB, SC(5), SH(5));

  // ---- graph path on segment features ----
  layer_kernel<32, false, false><<<nbSmall, 256, 0, stream>>>(
      XS, 32, Bn, 1, p_g2[0], 32, p_g2[1], nullptr, nullptr, C1, nullptr, ST(6));
  finalize_kernel<<<1, 32, 0, stream>>>(ST(6), p_g2[2], p_g2[3], fB, SC(6), SH(6));

  layer_kernel<32, false, false><<<nbSmall, 256, 0, stream>>>(
      C1, 32, Bn, 1, p_g2[4], 32, p_g2[5], SC(6), SH(6), C2, nullptr, ST(7));
  finalize_kernel<<<1, 32, 0, stream>>>(ST(7), p_g2[6], p_g2[7], fB, SC(7), SH(7));

  // ---- concat + output MLP ----
  concat_kernel<<<(Bn * 64 + 255) / 256, 256, 0, stream>>>(
      B2, C2, SC(5), SH(5), SC(7), SH(7), D0, Bn);

  layer_kernel<64, false, false><<<nbSmall, 256, 0, stream>>>(
      D0, 64, Bn, 1, p_ot[0], 64, p_ot[1], nullptr, nullptr, D1, nullptr, ST(8));
  finalize_kernel<<<1, 32, 0, stream>>>(ST(8), p_ot[2], p_ot[3], fB, SC(8), SH(8));

  layer_kernel<32, false, false><<<nbSmall, 256, 0, stream>>>(
      D1, 32, Bn, 1, p_ot[4], 32, p_ot[5], SC(8), SH(8), D2, nullptr, ST(9));
  finalize_kernel<<<1, 32, 0, stream>>>(ST(9), p_ot[6], p_ot[7], fB, SC(9), SH(9));

  final_kernel<<<(Bn * NL + 255) / 256, 256, 0, stream>>>(
      D2, SC(9), SH(9), Wo, bo, (float*)d_out, Bn, NL);
}
